// Head_22823456211418
// MI455X (gfx1250) — matrix operand order repaired and checked by reading, not compiled, not run on hardware
//
#include <hip/hip_runtime.h>
#include <stddef.h>

// ---------------------------------------------------------------- types
typedef __attribute__((ext_vector_type(16))) __bf16 v16bf;
typedef __attribute__((ext_vector_type(8)))  __bf16 v8bf;
typedef __attribute__((ext_vector_type(8)))  float  v8f;

#define WMMA_BF16(A, B, C) \
  __builtin_amdgcn_wmma_f32_16x16x32_bf16(false, (A), false, (B), (short)0, (C), false, false)

// Problem constants
#define BB  8
#define TT  2048
#define CC  1024
#define HS  64

__device__ __forceinline__ __bf16 f2bf(float f) {
  // round-to-nearest-even bf16
  unsigned u = __builtin_bit_cast(unsigned, f);
  unsigned r = u + 0x7FFFu + ((u >> 16) & 1u);
  unsigned short h = (unsigned short)(r >> 16);
  return __builtin_bit_cast(__bf16, h);
}

// ---------------------------------------------------------------- stage 0: f32 -> bf16
__global__ __launch_bounds__(256) void cvt_bf16_kernel(const float* __restrict__ in,
                                                       __bf16* __restrict__ out, int n8) {
  int i = blockIdx.x * 256 + threadIdx.x;
  if (i >= n8) return;
  const float* p = in + (size_t)i * 8;
  v8bf v;
#pragma unroll
  for (int j = 0; j < 8; ++j) v[j] = f2bf(p[j]);
  *(v8bf*)(out + (size_t)i * 8) = v;
}

// ---------------------------------------------------------------- stage 1: QKV GEMM + RoPE
// grid = (B*T/64, 3), block = 64 (2 waves). Wave w covers head-dims [32w, 32w+32).
// Each block computes 64 tokens for one of {q,k,v}; B-fragments reused over 4 M-tiles.
__global__ __launch_bounds__(64) void qkv_rope_kernel(const __bf16* __restrict__ xb,
                                                      const __bf16* __restrict__ Wb,
                                                      __bf16* __restrict__ qout,
                                                      __bf16* __restrict__ kout,
                                                      __bf16* __restrict__ vTout) {
  const int mat  = blockIdx.y;                       // 0=q 1=k 2=v
  const __bf16* W = Wb + (size_t)mat * (HS * CC);
  const int lane = threadIdx.x & 31;
  const int wv   = threadIdx.x >> 5;                 // 0..1
  const int l16  = lane & 15;
  const int lh   = lane >> 4;                        // 0..1
  const int tok0 = blockIdx.x * 64;
  const int ab   = lh * 16;                          // A K-chunk base
  const int bch  = lh * 16;                          // B K-chunk base
  const int h0   = wv * 32 + l16;                    // B frag columns

  v8f acc[4][2] = {};

  for (int kc = 0; kc < CC; kc += 32) {
    // B fragments (W rows are contraction-contiguous): 16 bf16 per lane, contiguous
    v16bf b0 = *(const v16bf*)(W + (size_t)h0 * CC + kc + bch);
    v16bf b1 = *(const v16bf*)(W + (size_t)(h0 + 16) * CC + kc + bch);
#pragma unroll
    for (int mi = 0; mi < 4; ++mi) {
      const __bf16* xr = xb + (size_t)(tok0 + mi * 16 + l16) * CC + kc + ab;
      v8bf a0 = *(const v8bf*)(xr);
      v8bf a1 = *(const v8bf*)(xr + 8);
      v16bf a = __builtin_shufflevector(a0, a1, 0,1,2,3,4,5,6,7,8,9,10,11,12,13,14,15);
      acc[mi][0] = WMMA_BF16(a, b0, acc[mi][0]);
      acc[mi][1] = WMMA_BF16(a, b1, acc[mi][1]);
    }
  }

  // Epilogue: RoPE for q/k (interleaved pairs live in adjacent lanes), vT store for v.
#pragma unroll
  for (int mi = 0; mi < 4; ++mi) {
#pragma unroll
    for (int n = 0; n < 2; ++n) {
      const int h = wv * 32 + n * 16 + l16;
#pragma unroll
      for (int i = 0; i < 8; ++i) {
        const int bt = tok0 + mi * 16 + lh * 8 + i;  // flat token
        float val = acc[mi][n][i];
        if (mat < 2) {
          float partner = __shfl_xor(val, 1, 32);    // h ^ 1 neighbor, same row
          const int t = bt & (TT - 1);
          // inv_freq = 10000^(-(h&~1)/64) ; ln(10000)=9.2103403719761836
          float invf = __expf((float)(h & ~1) * (-9.2103403719761836f / 64.0f));
          float ang = (float)t * invf;
          float sn, cs;
          __sincosf(ang, &sn, &cs);
          float r = (h & 1) ? fmaf(val, cs, partner * sn)
                            : fmaf(val, cs, -partner * sn);
          __bf16* dst = (mat == 0) ? qout : kout;
          dst[(size_t)bt * HS + h] = f2bf(r);
        } else {
          const int bb = bt >> 11, t = bt & (TT - 1);
          vTout[((size_t)bb * HS + h) * TT + t] = f2bf(val);
        }
      }
    }
  }
}

// ---------------------------------------------------------------- stage 2: flash attention
// grid = (T/16, B), block = 32 (one wave owns 16 query rows). Online softmax over
// 64-key blocks; 8 WMMA for Q.K^T + 8 WMMA for P.V per block.
__global__ __launch_bounds__(32) void attn_fwd_kernel(const __bf16* __restrict__ qg,
                                                      const __bf16* __restrict__ kg,
                                                      const __bf16* __restrict__ vTg,
                                                      float* __restrict__ out) {
  __shared__ __align__(32) __bf16 pbuf[16 * 72];     // P tile, 16 rows x 64 cols (stride 72)

  const int b     = blockIdx.y;
  const int qbase = blockIdx.x * 16;
  const int lane  = threadIdx.x;
  const int l16   = lane & 15;
  const int lh    = lane >> 4;
  const int ab    = lh * 16;

  // Q A-fragments (row = qbase + l16): two contiguous 8-elem chunks per lane
  const __bf16* qrow = qg + ((size_t)b * TT + qbase + l16) * HS;
  v8bf q00 = *(const v8bf*)(qrow + ab);
  v8bf q01 = *(const v8bf*)(qrow + 8 + ab);
  v8bf q10 = *(const v8bf*)(qrow + 32 + ab);
  v8bf q11 = *(const v8bf*)(qrow + 40 + ab);
  v16bf aq0 = __builtin_shufflevector(q00, q01, 0,1,2,3,4,5,6,7,8,9,10,11,12,13,14,15);
  v16bf aq1 = __builtin_shufflevector(q10, q11, 0,1,2,3,4,5,6,7,8,9,10,11,12,13,14,15);

  const float NEGINF = -__builtin_inff();
  float m[8], l[8];
  v8f o[4] = {};
#pragma unroll
  for (int i = 0; i < 8; ++i) { m[i] = NEGINF; l[i] = 0.0f; }

  const int kend = qbase + 16;                       // causal: last key needed = qbase+15
  for (int kb = 0; kb < kend; kb += 64) {
    if (kb + 64 < kend)                              // hint next K block (global_prefetch_b8)
      __builtin_prefetch(kg + ((size_t)b * TT + kb + 64 + lane) * HS, 0, 1);

    // ---- scores S = scale * Q K^T, causal-masked
    v8f s[4];
#pragma unroll
    for (int n = 0; n < 4; ++n) {
      const __bf16* krow = kg + ((size_t)b * TT + kb + n * 16 + l16) * HS + lh * 16;
      v16bf bk0 = *(const v16bf*)(krow);
      v16bf bk1 = *(const v16bf*)(krow + 32);
      v8f c = {};
      c = WMMA_BF16(aq0, bk0, c);
      c = WMMA_BF16(aq1, bk1, c);
#pragma unroll
      for (int i = 0; i < 8; ++i) {
        const int row = qbase + lh * 8 + i;
        const int col = kb + n * 16 + l16;
        s[n][i] = (col <= row) ? c[i] * 0.125f : NEGINF;   // 1/sqrt(64)
      }
    }

    // ---- online softmax (rows span 16 lanes of a half-wave)
    float alpha[8];
#pragma unroll
    for (int i = 0; i < 8; ++i) {
      float mx = fmaxf(fmaxf(s[0][i], s[1][i]), fmaxf(s[2][i], s[3][i]));
      mx = fmaxf(mx, __shfl_xor(mx, 1, 32));
      mx = fmaxf(mx, __shfl_xor(mx, 2, 32));
      mx = fmaxf(mx, __shfl_xor(mx, 4, 32));
      mx = fmaxf(mx, __shfl_xor(mx, 8, 32));
      float mnew = fmaxf(m[i], mx);
      alpha[i] = __expf(m[i] - mnew);
      m[i] = mnew;
      float rs = 0.0f;
#pragma unroll
      for (int n = 0; n < 4; ++n) {
        float p = __expf(s[n][i] - mnew);
        s[n][i] = p;
        rs += p;
      }
      rs += __shfl_xor(rs, 1, 32);
      rs += __shfl_xor(rs, 2, 32);
      rs += __shfl_xor(rs, 4, 32);
      rs += __shfl_xor(rs, 8, 32);
      l[i] = l[i] * alpha[i] + rs;
#pragma unroll
      for (int n = 0; n < 4; ++n) o[n][i] *= alpha[i];
    }

    // ---- P (C-layout) -> LDS -> A-fragments (bf16)
#pragma unroll
    for (int n = 0; n < 4; ++n)
#pragma unroll
      for (int i = 0; i < 8; ++i)
        pbuf[(lh * 8 + i) * 72 + n * 16 + l16] = f2bf(s[n][i]);
    // same-wave DS ordering: compiler inserts s_wait_dscnt on the dependent loads
    const __bf16* pr = &pbuf[l16 * 72 + ab];
    v8bf p00 = *(const v8bf*)(pr);
    v8bf p01 = *(const v8bf*)(pr + 8);
    v8bf p10 = *(const v8bf*)(pr + 32);
    v8bf p11 = *(const v8bf*)(pr + 40);
    v16bf pa0 = __builtin_shufflevector(p00, p01, 0,1,2,3,4,5,6,7,8,9,10,11,12,13,14,15);
    v16bf pa1 = __builtin_shufflevector(p10, p11, 0,1,2,3,4,5,6,7,8,9,10,11,12,13,14,15);

    // ---- O += P V  (V transposed => contiguous per-lane B chunks)
    const __bf16* vb = vTg + (size_t)b * HS * TT;
#pragma unroll
    for (int n = 0; n < 4; ++n) {
      const __bf16* vr = vb + (size_t)(n * 16 + l16) * TT + kb + lh * 16;
      v16bf bv0 = *(const v16bf*)(vr);
      v16bf bv1 = *(const v16bf*)(vr + 32);
      o[n] = WMMA_BF16(pa0, bv0, o[n]);
      o[n] = WMMA_BF16(pa1, bv1, o[n]);
    }
  }

  // ---- epilogue: normalize and store fp32
#pragma unroll
  for (int n = 0; n < 4; ++n)
#pragma unroll
    for (int i = 0; i < 8; ++i) {
      const int row = qbase + lh * 8 + i;
      const int h   = n * 16 + l16;
      out[((size_t)b * TT + row) * HS + h] = o[n][i] / l[i];
    }
}

// ---------------------------------------------------------------- launch
extern "C" void kernel_launch(void* const* d_in, const int* in_sizes, int n_in,
                              void* d_out, int out_size, void* d_ws, size_t ws_size,
                              hipStream_t stream) {
  (void)in_sizes; (void)n_in; (void)out_size; (void)ws_size;
  const float* x  = (const float*)d_in[0];   // (8, 2048, 1024)
  const float* Wq = (const float*)d_in[1];   // (64, 1024)
  const float* Wk = (const float*)d_in[2];
  const float* Wv = (const float*)d_in[3];
  float* out = (float*)d_out;                // (8, 2048, 64)

  // workspace layout (bf16 elements); all 32B aligned
  const size_t XN = (size_t)BB * TT * CC;    // 16,777,216
  const size_t WN = (size_t)HS * CC;         // 65,536
  const size_t QN = (size_t)BB * TT * HS;    // 1,048,576
  __bf16* xb = (__bf16*)d_ws;
  __bf16* Wb = xb + XN;                      // 3 matrices, concat
  __bf16* qb = Wb + 3 * WN;
  __bf16* kb = qb + QN;
  __bf16* vT = kb + QN;

  // stage 0: down-convert once (keeps GEMM loops free of cvt VALU work)
  cvt_bf16_kernel<<<dim3((unsigned)(XN / 8 / 256)), 256, 0, stream>>>(x, xb, (int)(XN / 8));
  cvt_bf16_kernel<<<dim3((unsigned)(WN / 8 / 256)), 256, 0, stream>>>(Wq, Wb, (int)(WN / 8));
  cvt_bf16_kernel<<<dim3((unsigned)(WN / 8 / 256)), 256, 0, stream>>>(Wk, Wb + WN, (int)(WN / 8));
  cvt_bf16_kernel<<<dim3((unsigned)(WN / 8 / 256)), 256, 0, stream>>>(Wv, Wb + 2 * WN, (int)(WN / 8));

  // stage 1: QKV projection + RoPE (q,k row-major bf16; v transposed bf16)
  qkv_rope_kernel<<<dim3(BB * TT / 64, 3), 64, 0, stream>>>(xb, Wb, qb, kb, vT);

  // stage 2: causal flash attention
  attn_fwd_kernel<<<dim3(TT / 16, BB), 32, 0, stream>>>(qb, kb, vT, out);
}
